// QuantizedLinear_30803505447137
// MI455X (gfx1250) — matrix operand order repaired and checked by reading, not compiled, not run on hardware
//
#include <hip/hip_runtime.h>

// ---------------------------------------------------------------------------
// QuantizedLinear for MI455X (gfx1250, wave32, WMMA):
//   out[M=8192, N=4096] = x[M, K=4096] @ W[N, K]^T + bias
// Compute-bound (AI ~965 FLOP/B) -> v_wmma_f32_16x16x32_bf16, f32 accumulate.
//   Pass 1a: dequantize W (codebook gather + scales) -> bf16 in ws[0..32MiB)
//   Pass 1b: convert x -> bf16 in ws[32MiB..96MiB)   (x stays L2-resident)
//   Pass 2 : tiled WMMA GEMM, async global->LDS staging (ASYNCcnt),
//            double-buffered LDS, no inner-loop conversion, no spills.
// ---------------------------------------------------------------------------

typedef __attribute__((ext_vector_type(16))) __bf16 v16bf;
typedef __attribute__((ext_vector_type(8)))  float  v8f;

#define M_TOTAL 8192   // BATCH * SEQ
#define N_OUT   4096
#define K_IN    4096
#define BM      128
#define BN      128
#define BK      64
#define LDT     72     // LDS row stride in bf16 elems (64 + 8 pad, 16B-aligned rows)
#define NK      (K_IN / BK)   // 64 staged K-steps (2 WMMA sub-steps each)

#if __has_builtin(__builtin_amdgcn_global_load_async_to_lds_b128)
#define USE_ASYNC 1
#else
#define USE_ASYNC 0
#endif

// Builtin prototype (from hipcc diagnostic): 16-byte int vector pointers,
// global (AS1) source and LDS (AS3) destination, + imm offset / cpol.
typedef int v4i_vs __attribute__((vector_size(16)));
typedef v4i_vs __attribute__((address_space(1)))* gp128_t;
typedef v4i_vs __attribute__((address_space(3)))* lp128_t;

__device__ __forceinline__ void cp16_g2l(const void* g, void* l) {
#if USE_ASYNC
  // CDNA5 async copy: GLOBAL_LOAD_ASYNC_TO_LDS_B128, tracked by ASYNCcnt
  __builtin_amdgcn_global_load_async_to_lds_b128((gp128_t)g, (lp128_t)l, 0, 0);
#else
  *(uint4*)l = *(const uint4*)g;
#endif
}

__device__ __forceinline__ void wait_async0() {
#if USE_ASYNC
#if __has_builtin(__builtin_amdgcn_s_wait_asynccnt)
  __builtin_amdgcn_s_wait_asynccnt(0);
#else
  asm volatile("s_wait_asynccnt 0x0" ::: "memory");
#endif
#endif
}

__device__ __forceinline__ unsigned short f2bf(float f) {
  // round-to-nearest-even f32 -> bf16
  unsigned u = __float_as_uint(f);
  return (unsigned short)((u + 0x7FFFu + ((u >> 16) & 1u)) >> 16);
}

// ---------------------------------------------------------------------------
// Pass 1a: W[o, 4j..4j+3] = codebook[assignments[o,j]] * scale_out[o] * scale_in
// Scales folded in fp32 before bf16 rounding; codebook (64KB) is L2-resident.
// ---------------------------------------------------------------------------
__global__ __launch_bounds__(256) void dequant_kernel(
    const int* __restrict__ assignments,
    const float* __restrict__ codebook,
    const float* __restrict__ scale_out,
    const float* __restrict__ scale_in,
    unsigned short* __restrict__ wb) {
  int gid = blockIdx.x * 256 + threadIdx.x;   // 0 .. N_OUT*(K_IN/4)-1
  int o = gid >> 10;
  int j = gid & 1023;
  int idx = assignments[gid];
  float4 c  = ((const float4*)codebook)[idx];
  float  so = scale_out[o];
  float4 si = ((const float4*)scale_in)[j];
  ushort4 r;
  r.x = f2bf(c.x * so * si.x);
  r.y = f2bf(c.y * so * si.y);
  r.z = f2bf(c.z * so * si.z);
  r.w = f2bf(c.w * so * si.w);
  ((ushort4*)wb)[gid] = r;                    // row-major W bf16 [N_OUT, K_IN]
}

// ---------------------------------------------------------------------------
// Pass 1b: x f32 -> bf16 (8 elements per thread; pure bandwidth)
// ---------------------------------------------------------------------------
__global__ __launch_bounds__(256) void xcvt_kernel(
    const float* __restrict__ x, unsigned short* __restrict__ xb) {
  int gid = blockIdx.x * 256 + threadIdx.x;   // 0 .. M*K/8-1
  const float4* p = (const float4*)x + gid * 2;
  float4 a = p[0], b = p[1];
  ushort4 lo, hi;
  lo.x = f2bf(a.x); lo.y = f2bf(a.y); lo.z = f2bf(a.z); lo.w = f2bf(a.w);
  hi.x = f2bf(b.x); hi.y = f2bf(b.y); hi.z = f2bf(b.z); hi.w = f2bf(b.w);
  ((ushort4*)xb)[gid * 2]     = lo;
  ((ushort4*)xb)[gid * 2 + 1] = hi;
}

// ---------------------------------------------------------------------------
// Pass 2: tiled GEMM. 256 threads = 8 wave32s. Block tile 128x128x64.
// Wave tile 32x64 -> 2x4 v8f accumulators, 16 WMMAs per staged K-step.
// Async global->LDS double buffering; s_wait_asynccnt + barrier per step.
// ---------------------------------------------------------------------------
__global__ __launch_bounds__(256) void gemm_kernel(
    const unsigned short* __restrict__ xb,  // [M_TOTAL, K_IN] bf16
    const unsigned short* __restrict__ wb,  // [N_OUT,  K_IN] bf16
    const float* __restrict__ bias,         // [N_OUT]
    float* __restrict__ out) {              // [M_TOTAL, N_OUT] f32
  __shared__ unsigned short ldsA[2][BM * LDT];
  __shared__ unsigned short ldsB[2][BN * LDT];

  const int tid  = threadIdx.x;
  const int lane = tid & 31;
  const int wave = tid >> 5;
  const int wm   = wave & 3;     // 4 waves along M
  const int wn   = wave >> 2;    // 2 waves along N
  const int m0   = blockIdx.y * BM;
  const int n0   = blockIdx.x * BN;

  // Stage one 128x64 bf16 tile pair: 1024 16B-chunks each, 4+4 per thread.
  auto stage = [&](int buf, int kt) {
    const unsigned short* xa = xb + m0 * K_IN + kt * BK;
    const unsigned short* wa = wb + n0 * K_IN + kt * BK;
#pragma unroll
    for (int c = 0; c < 4; ++c) {
      int idx = tid + c * 256;
      int row = idx >> 3, col = (idx & 7) * 8;   // 8 chunks of 8 bf16 per row
      cp16_g2l(xa + row * K_IN + col, &ldsA[buf][row * LDT + col]);
    }
#pragma unroll
    for (int c = 0; c < 4; ++c) {
      int idx = tid + c * 256;
      int row = idx >> 3, col = (idx & 7) * 8;
      cp16_g2l(wa + row * K_IN + col, &ldsB[buf][row * LDT + col]);
    }
  };

  const v8f vzero = {};
  v8f acc[2][4];
#pragma unroll
  for (int mi = 0; mi < 2; ++mi)
#pragma unroll
    for (int ni = 0; ni < 4; ++ni) acc[mi][ni] = vzero;

  union AB { v16bf v; uint4 u[2]; };

  const int r16 = lane & 15;
  const int kc  = (lane >> 4) * 8;    // A K-chunk base per ISA VGPR layout
  const int ko  = (lane >> 4) * 16;   // unused; B reads K at kc like A

  auto compute = [&](int buf) {
#pragma unroll
    for (int kk = 0; kk < 2; ++kk) {  // two 16x16x32 sub-steps per staged tile
      const int kb = kk * 32;
      AB af[2], bfr[4];
#pragma unroll
      for (int mi = 0; mi < 2; ++mi) {
        // A 16x32: lane holds row M=lane&15, K = [kc,kc+8) and [kc+16,kc+24)
        const unsigned short* p = &ldsA[buf][(wm * 32 + mi * 16 + r16) * LDT + kb];
        af[mi].u[0] = *(const uint4*)(p + kc);
        af[mi].u[1] = *(const uint4*)(p + kc + 16);
      }
#pragma unroll
      for (int ni = 0; ni < 4; ++ni) {
        // B 32x16: lane holds column N=lane&15 (a W row), K = [kc,kc+8) and [kc+16,kc+24)
        const unsigned short* p = &ldsB[buf][(wn * 64 + ni * 16 + r16) * LDT + kb + kc];
        bfr[ni].u[0] = *(const uint4*)(p);
        bfr[ni].u[1] = *(const uint4*)(p + 16);
      }
#pragma unroll
      for (int mi = 0; mi < 2; ++mi)
#pragma unroll
        for (int ni = 0; ni < 4; ++ni)
          acc[mi][ni] = __builtin_amdgcn_wmma_f32_16x16x32_bf16(
              false, af[mi].v, false, bfr[ni].v, (short)0, acc[mi][ni],
              false, false);
    }
  };

  stage(0, 0);
  wait_async0();
  __syncthreads();

#pragma unroll 1
  for (int kt = 0; kt < NK; ++kt) {
    const int cur = kt & 1;
    if (kt + 1 < NK) stage(cur ^ 1, kt + 1);  // async prefetch overlaps compute
    compute(cur);
    wait_async0();                            // prefetch landed in LDS
    __syncthreads();
  }

  // Epilogue: C/D 16x16 f32 layout -> lane holds col N=lane&15,
  // VGPR r holds row M = r + (lane>>4)*8. Add bias, store f32.
#pragma unroll
  for (int ni = 0; ni < 4; ++ni) {
    const int ncol = n0 + wn * 64 + ni * 16 + r16;
    const float b = bias[ncol];
#pragma unroll
    for (int mi = 0; mi < 2; ++mi) {
      const int mbase = m0 + wm * 32 + mi * 16 + (lane >> 4) * 8;
#pragma unroll
      for (int r = 0; r < 8; ++r) {
        out[(mbase + r) * N_OUT + ncol] = acc[mi][ni][r] + b;
      }
    }
  }
}

// ---------------------------------------------------------------------------
extern "C" void kernel_launch(void* const* d_in, const int* in_sizes, int n_in,
                              void* d_out, int out_size, void* d_ws, size_t ws_size,
                              hipStream_t stream) {
  const float* x           = (const float*)d_in[0];
  const float* codebook    = (const float*)d_in[1];
  const int*   assignments = (const int*)d_in[2];
  const float* scale_out   = (const float*)d_in[3];
  const float* scale_in    = (const float*)d_in[4];
  const float* bias        = (const float*)d_in[5];

  // ws layout: [0, 32MiB) bf16 W ; [32MiB, 96MiB) bf16 x
  unsigned short* wb = (unsigned short*)d_ws;
  unsigned short* xbuf = wb + (size_t)N_OUT * K_IN;   // +16M elems = +32MiB

  const int n_assign = N_OUT * (K_IN / 4);            // 4M threads
  dequant_kernel<<<n_assign / 256, 256, 0, stream>>>(
      assignments, codebook, scale_out, scale_in, wb);

  const int n_x8 = (M_TOTAL * K_IN) / 8;              // 4M threads
  xcvt_kernel<<<n_x8 / 256, 256, 0, stream>>>(x, xbuf);

  dim3 grid(N_OUT / BN, M_TOTAL / BM);                // (32, 64)
  gemm_kernel<<<grid, 256, 0, stream>>>(xbuf, wb, bias, (float*)d_out);
}
